// MultiHeadAttention_13649406067023
// MI455X (gfx1250) — compile-verified, matrix operand order checked by reading, not run on hardware
//
#include <hip/hip_runtime.h>
#include <hip/hip_bf16.h>

typedef __bf16 bf16;
typedef __attribute__((ext_vector_type(16))) __bf16 v16bf;
typedef __attribute__((ext_vector_type(8)))  float  v8f;
typedef __attribute__((ext_vector_type(4)))  int    v4i;

#ifndef __has_builtin
#define __has_builtin(x) 0
#endif
#if __has_builtin(__builtin_amdgcn_global_load_async_to_lds_b128)
#define HAVE_ASYNC 1
#else
#define HAVE_ASYNC 0
#endif

#define AS1 __attribute__((address_space(1)))
#define AS3 __attribute__((address_space(3)))

union FragBF { v16bf v; uint4 q[2]; };

__device__ __forceinline__ bf16 f2bf(float f) {
  unsigned u = __builtin_bit_cast(unsigned, f);
  unsigned short s = (unsigned short)((u + 0x7FFFu + ((u >> 16) & 1u)) >> 16); // RNE
  return __builtin_bit_cast(bf16, s);
}

__device__ __forceinline__ v8f wmma_bf16(const FragBF& a, const FragBF& b, v8f c) {
  // (neg_a, A, neg_b, B, c_mod, C, reuse_a, reuse_b)
  return __builtin_amdgcn_wmma_f32_16x16x32_bf16(false, a.v, false, b.v, (short)0, c, false, false);
}

// 16-byte global -> LDS copy; async DMA (ASYNCcnt) when the toolchain has it.
__device__ __forceinline__ void cp16_g2l(const bf16* g, bf16* l) {
#if HAVE_ASYNC
  __builtin_amdgcn_global_load_async_to_lds_b128(
      (AS1 v4i*)g, (AS3 v4i*)l, 0, 0);
#else
  *(uint4*)l = *(const uint4*)g;
#endif
}

__device__ __forceinline__ void wait_async0() {
#if HAVE_ASYNC
#if __has_builtin(__builtin_amdgcn_s_wait_asynccnt)
  __builtin_amdgcn_s_wait_asynccnt(0);
#else
  asm volatile("s_wait_asynccnt 0" ::: "memory");
#endif
#endif
}

// ---------------- conversion kernels ----------------

__global__ void k_cvt_bf16(const float* __restrict__ in, bf16* __restrict__ out) {
  const int i = (blockIdx.x * 256 + threadIdx.x) * 4;
  float4 v = *(const float4*)(in + i);
  out[i + 0] = f2bf(v.x);
  out[i + 1] = f2bf(v.y);
  out[i + 2] = f2bf(v.z);
  out[i + 3] = f2bf(v.w);
}

// W: [K, Ncols] fp32 row-major  ->  Wt: [Ncols, K] bf16 row-major (i.e. B^T)
__global__ void k_tcvt_bf16(const float* __restrict__ W, bf16* __restrict__ Wt,
                            int K, int Ncols) {
  const long i = (long)blockIdx.x * 256 + threadIdx.x;
  const int n = (int)(i / K);
  const int k = (int)(i % K);
  Wt[i] = f2bf(W[(size_t)k * Ncols + n]);
}

// ---------------- GEMM: C[M,N] = A[M,K] * Bt[N,K]^T + bias ----------------
// block = 256 threads (8 waves), block tile 128M x 128N, wave tile 32x64
// (waves 4x2). BK=32, double-buffered LDS filled by async-to-LDS DMA.

template <bool OBF>
__global__ void __launch_bounds__(256) k_gemm(const bf16* __restrict__ A,
                                              const bf16* __restrict__ Bt,
                                              const float* __restrict__ bias,
                                              void* __restrict__ Cout,
                                              int Ncols, int K) {
  __shared__ bf16 sA[2][128][40];  // 40*2B = 80B pitch (16B-aligned rows)
  __shared__ bf16 sB[2][128][40];

  const int tid  = threadIdx.x;
  const int wave = tid >> 5, lane = tid & 31;
  const int lrow = lane & 15, hlf = lane >> 4, kbo = hlf << 3;
  const int m0 = blockIdx.x * 128, n0 = blockIdx.y * 128;
  const int mw = (wave >> 1) * 32, nw = (wave & 1) * 64;

  v8f acc[2][4];
#pragma unroll
  for (int i = 0; i < 2; ++i)
#pragma unroll
    for (int j = 0; j < 4; ++j)
#pragma unroll
      for (int e = 0; e < 8; ++e) acc[i][j][e] = 0.f;

  // staging: 1024 x 16B chunks per K-step (512 A + 512 B), 4 per thread.
  const int ar0 = tid >> 2,          ac0 = (tid & 3) * 8;
  const int ar1 = (tid + 256) >> 2,  ac1 = ac0;  // rows 64..127, same col chunk

  auto stage = [&](int buf, int ko) {
    cp16_g2l(A  + (size_t)(m0 + ar0) * K + ko + ac0, &sA[buf][ar0][ac0]);
    cp16_g2l(A  + (size_t)(m0 + ar1) * K + ko + ac1, &sA[buf][ar1][ac1]);
    cp16_g2l(Bt + (size_t)(n0 + ar0) * K + ko + ac0, &sB[buf][ar0][ac0]);
    cp16_g2l(Bt + (size_t)(n0 + ar1) * K + ko + ac1, &sB[buf][ar1][ac1]);
  };

  stage(0, 0);
  wait_async0();
  __syncthreads();

  int buf = 0;
  for (int ko = 0; ko < K; ko += 32) {
    if (ko + 32 < K) stage(buf ^ 1, ko + 32);  // DMA next tile during compute

    FragBF fa[2], fb[4];
#pragma unroll
    for (int i = 0; i < 2; ++i) {
      const bf16* ap = &sA[buf][mw + i * 16 + lrow][kbo];
      fa[i].q[0] = *(const uint4*)ap;
      fa[i].q[1] = *(const uint4*)(ap + 16);
    }
#pragma unroll
    for (int j = 0; j < 4; ++j) {
      const bf16* bp = &sB[buf][nw + j * 16 + lrow][kbo];
      fb[j].q[0] = *(const uint4*)bp;
      fb[j].q[1] = *(const uint4*)(bp + 16);
    }
#pragma unroll
    for (int i = 0; i < 2; ++i)
#pragma unroll
      for (int j = 0; j < 4; ++j)
        acc[i][j] = wmma_bf16(fa[i], fb[j], acc[i][j]);

    wait_async0();     // my async fills of buf^1 landed
    __syncthreads();   // everyone done reading buf / writing buf^1
    buf ^= 1;
  }

#pragma unroll
  for (int j = 0; j < 4; ++j) {
    const int col = n0 + nw + j * 16 + lrow;
    const float bv = bias[col];
#pragma unroll
    for (int i = 0; i < 2; ++i) {
#pragma unroll
      for (int e = 0; e < 8; ++e) {
        const int row = m0 + mw + i * 16 + e + hlf * 8;
        const float v = acc[i][j][e] + bv;
        if constexpr (OBF)
          ((bf16*)Cout)[(size_t)row * Ncols + col] = f2bf(v);
        else
          ((float*)Cout)[(size_t)row * Ncols + col] = v;
      }
    }
  }
}

// ---------------- flash attention ----------------
// qkv: [B*N, 3D] bf16, row n holds [Q | K | V], head h at column h*64.
// grid = (N/128, H, B), block = 256 (8 waves, 16 q-rows per wave).

__global__ void __launch_bounds__(256) k_attn(const bf16* __restrict__ qkv,
                                              bf16* __restrict__ ctx) {
  __shared__ bf16 sVt[64][136];      // V^T tile (hd x kv), 136*2B pitch
  __shared__ bf16 sP[8][16][136];    // per-wave P tile (16 q x 128 kv)

  const int qt = blockIdx.x, h = blockIdx.y, b = blockIdx.z;
  const int tid  = threadIdx.x;
  const int wave = tid >> 5, lane = tid & 31;
  const int lrow = lane & 15, hlf = lane >> 4, kbo = hlf << 3;

  const size_t rs = 3072;  // qkv row stride
  const bf16* Qb = qkv + (size_t)b * 2048 * rs + h * 64;
  const bf16* Kb = Qb + 1024;
  const bf16* Vb = Qb + 2048;

  const int q0 = qt * 128 + wave * 16;

  // Q fragments for the wave's 16 rows (K = 64 -> 2 ksteps of 32)
  FragBF fq[2];
  {
    const bf16* qp = Qb + (size_t)(q0 + lrow) * rs;
#pragma unroll
    for (int ks = 0; ks < 2; ++ks) {
      fq[ks].q[0] = *(const uint4*)(qp + ks * 32 + kbo);
      fq[ks].q[1] = *(const uint4*)(qp + ks * 32 + kbo + 16);
    }
  }

  float m_i[8], l_i[8];
  v8f accO[4];
#pragma unroll
  for (int i = 0; i < 8; ++i) { m_i[i] = -1e30f; l_i[i] = 0.f; }
#pragma unroll
  for (int t = 0; t < 4; ++t)
#pragma unroll
    for (int e = 0; e < 8; ++e) accO[t][e] = 0.f;

  const float scale = 0.125f;  // 64^-0.5

  for (int kv0 = 0; kv0 < 2048; kv0 += 128) {
    __syncthreads();
    // stage V^T into LDS (transposed so PV B-fragments are contiguous)
    for (int i = tid; i < 128 * 64; i += 256) {
      const int n = i >> 6, d = i & 63;
      sVt[d][n] = Vb[(size_t)(kv0 + n) * rs + d];
    }
    __syncthreads();

    // S = Q * K^T : K rows in memory are already B^T [kv x hd]
    v8f accS[8];
#pragma unroll
    for (int nt = 0; nt < 8; ++nt) {
#pragma unroll
      for (int e = 0; e < 8; ++e) accS[nt][e] = 0.f;
      const bf16* kp = Kb + (size_t)(kv0 + nt * 16 + lrow) * rs;
      FragBF fk0, fk1;
      fk0.q[0] = *(const uint4*)(kp + kbo);
      fk0.q[1] = *(const uint4*)(kp + kbo + 16);
      fk1.q[0] = *(const uint4*)(kp + 32 + kbo);
      fk1.q[1] = *(const uint4*)(kp + 32 + kbo + 16);
      accS[nt] = wmma_bf16(fq[0], fk0, accS[nt]);
      accS[nt] = wmma_bf16(fq[1], fk1, accS[nt]);
    }

    // online softmax; C-layout: vgpr e, half -> row e+8*hlf, col lrow
#pragma unroll
    for (int i = 0; i < 8; ++i) {
      float mx = -1e30f;
#pragma unroll
      for (int nt = 0; nt < 8; ++nt) mx = fmaxf(mx, accS[nt][i]);
      mx = fmaxf(mx, __shfl_xor(mx, 1));
      mx = fmaxf(mx, __shfl_xor(mx, 2));
      mx = fmaxf(mx, __shfl_xor(mx, 4));
      mx = fmaxf(mx, __shfl_xor(mx, 8));
      const float mnew  = fmaxf(m_i[i], mx * scale);
      const float alpha = __expf(m_i[i] - mnew);
      m_i[i] = mnew;
      float rsum = 0.f;
#pragma unroll
      for (int nt = 0; nt < 8; ++nt) {
        const float p = __expf(accS[nt][i] * scale - mnew);
        rsum += p;
        sP[wave][i + hlf * 8][nt * 16 + lrow] = f2bf(p);
      }
      rsum += __shfl_xor(rsum, 1);
      rsum += __shfl_xor(rsum, 2);
      rsum += __shfl_xor(rsum, 4);
      rsum += __shfl_xor(rsum, 8);
      l_i[i] = l_i[i] * alpha + rsum;
#pragma unroll
      for (int t = 0; t < 4; ++t) accO[t][i] *= alpha;
    }
    __syncthreads();

    // O += P * V  (K = 128 kv -> 4 ksteps)
#pragma unroll
    for (int ks = 0; ks < 4; ++ks) {
      FragBF fp;
      const bf16* pp = &sP[wave][lrow][ks * 32 + kbo];
      fp.q[0] = *(const uint4*)pp;
      fp.q[1] = *(const uint4*)(pp + 16);
#pragma unroll
      for (int t = 0; t < 4; ++t) {
        FragBF fv;
        const bf16* vp = &sVt[t * 16 + lrow][ks * 32 + kbo];
        fv.q[0] = *(const uint4*)vp;
        fv.q[1] = *(const uint4*)(vp + 16);
        accO[t] = wmma_bf16(fp, fv, accO[t]);
      }
    }
  }

  // normalize and write context in [B*N, D] layout (undoes head transpose)
  bf16* cb = ctx + (size_t)(b * 2048 + q0) * 1024 + h * 64;
#pragma unroll
  for (int i = 0; i < 8; ++i) {
    const float inv = 1.0f / l_i[i];
    const int r = i + hlf * 8;
#pragma unroll
    for (int t = 0; t < 4; ++t)
      cb[(size_t)r * 1024 + t * 16 + lrow] = f2bf(accO[t][i] * inv);
  }
}

// ---------------- launch ----------------

extern "C" void kernel_launch(void* const* d_in, const int* in_sizes, int n_in,
                              void* d_out, int out_size, void* d_ws, size_t ws_size,
                              hipStream_t stream) {
  (void)in_sizes; (void)n_in; (void)out_size; (void)ws_size;
  const float* x      = (const float*)d_in[0];
  const float* w_qkv  = (const float*)d_in[1];
  const float* b_qkv  = (const float*)d_in[2];
  const float* w_proj = (const float*)d_in[3];
  const float* b_proj = (const float*)d_in[4];
  float* out = (float*)d_out;

  char* p = (char*)d_ws;
  bf16* x_bf   = (bf16*)p; p += (size_t)8192 * 1024 * 2;  // x in bf16
  bf16* wqkvT  = (bf16*)p; p += (size_t)3072 * 1024 * 2;  // w_qkv^T
  bf16* wprojT = (bf16*)p; p += (size_t)1024 * 1024 * 2;  // w_proj^T
  bf16* qkv    = (bf16*)p; p += (size_t)8192 * 3072 * 2;  // qkv activations
  bf16* ctx    = (bf16*)p; p += (size_t)8192 * 1024 * 2;  // attention output

  // 1) precision conversion / weight transposes
  k_cvt_bf16<<<8192, 256, 0, stream>>>(x, x_bf);
  k_tcvt_bf16<<<12288, 256, 0, stream>>>(w_qkv, wqkvT, 1024, 3072);
  k_tcvt_bf16<<<4096, 256, 0, stream>>>(w_proj, wprojT, 1024, 1024);

  // 2) QKV projection: [8192,1024] x [1024,3072] -> bf16 qkv
  k_gemm<true><<<dim3(64, 24), 256, 0, stream>>>(x_bf, wqkvT, b_qkv, qkv, 3072, 1024);

  // 3) flash attention per (q-tile, head, batch)
  k_attn<<<dim3(16, 16, 4), 256, 0, stream>>>(qkv, ctx);

  // 4) output projection: [8192,1024] x [1024,1024] -> fp32 out
  k_gemm<false><<<dim3(64, 8), 256, 0, stream>>>(ctx, wprojT, b_proj, out, 1024, 1024);
}
